// MultiheadSelfAttention_22754736734479
// MI455X (gfx1250) — matrix operand order repaired and checked by reading, not compiled, not run on hardware
//
#include <hip/hip_runtime.h>

typedef __bf16 bf16;
typedef __attribute__((ext_vector_type(8)))  __bf16 v8bf;
typedef __attribute__((ext_vector_type(16))) __bf16 v16bf;
typedef __attribute__((ext_vector_type(8)))  float  v8f;
typedef __attribute__((ext_vector_type(4)))  unsigned u32x4;
typedef __attribute__((ext_vector_type(8)))  unsigned u32x8;

#define D_MODEL 1024
#define SEQ     2048
#define NBATCH  2
#define NHEADS  16
#define DK      64
#define MTOT    (NBATCH*SEQ)   // 4096 tokens

// ---------------------------------------------------------------------------
// helpers
// ---------------------------------------------------------------------------
__device__ __forceinline__ v16bf cat8(v8bf lo, v8bf hi) {
  return __builtin_shufflevector(lo, hi, 0,1,2,3,4,5,6,7,8,9,10,11,12,13,14,15);
}

__device__ __forceinline__ v8f wmma_bf16(v16bf a, v16bf b, v8f c) {
  // D = A(16x32 bf16) * B(32x16 bf16) + C(16x16 f32)
  return __builtin_amdgcn_wmma_f32_16x16x32_bf16(false, a, false, b, (short)0, c,
                                                 false, false);
}

__device__ __forceinline__ unsigned lds_off(const void* p) {
  // LDS flat address = {SHARED_BASE, byte offset}; low 32 bits are the offset.
  return (unsigned)(uintptr_t)p;
}

// Async global -> LDS copy, 16 bytes per lane.  Tracked with ASYNCcnt.
__device__ __forceinline__ void async_ld_b128(unsigned ldsaddr, const void* gptr) {
  asm volatile("global_load_async_to_lds_b128 %0, %1, off"
               :: "v"(ldsaddr), "v"(gptr) : "memory");
}
__device__ __forceinline__ void wait_asynccnt0() {
  asm volatile("s_wait_asynccnt 0x0" ::: "memory");
}

// Tensor Data Mover: 2D tile load described by D# groups 0 (4 SGPRs) and
// 1 (8 SGPRs); VADDR2/3 = NULL -> 2D tensor.  Tracked with TENSORcnt.
__device__ __forceinline__ void tdm_load_2d(u32x4 g0, u32x8 g1) {
  asm volatile("tensor_load_to_lds %0, %1" :: "s"(g0), "s"(g1) : "memory");
}
__device__ __forceinline__ void wait_tensorcnt0() {
  asm volatile("s_wait_tensorcnt 0x0" ::: "memory");
}

// Build D# group 0: count=1, lds byte addr, 57-bit global byte addr, type=2.
__device__ __forceinline__ u32x4 tdm_group0(unsigned ldsaddr, const void* g) {
  unsigned long long ga = (unsigned long long)(uintptr_t)g;
  u32x4 r = { 1u, ldsaddr, (unsigned)ga,
              (unsigned)((ga >> 32) & 0x01FFFFFFu) | (2u << 30) };
  return r;
}

// Build D# group 1 for a 2D tile (data_size = 2 bytes), with LDS padding.
// pad_interval code: pad after 2^(code+1) DWORDs; pad_amount code: (code+1) DWORDs.
__device__ __forceinline__ u32x8 tdm_group1(unsigned tensor_d0, unsigned tensor_d1,
                                            unsigned tile_d0, unsigned tile_d1,
                                            unsigned long long stride0,
                                            unsigned pad_interval, unsigned pad_amount) {
  u32x8 r;
  r[0] = (1u << 16) | (1u << 20) | (pad_interval << 22) | (pad_amount << 25);
  r[1] = (tensor_d0 & 0xFFFFu) << 16;                               // dim0[15:0]
  r[2] = ((tensor_d0 >> 16) & 0xFFFFu) | ((tensor_d1 & 0xFFFFu) << 16);
  r[3] = ((tensor_d1 >> 16) & 0xFFFFu) | ((tile_d0 & 0xFFFFu) << 16);
  r[4] = (tile_d1 & 0xFFFFu);                                       // tile_dim2 = 0
  r[5] = (unsigned)(stride0 & 0xFFFFFFFFull);
  r[6] = (unsigned)((stride0 >> 32) & 0xFFFFull);                   // dim1_stride = 0
  r[7] = 0u;
  return r;
}

// ---------------------------------------------------------------------------
// fp32 -> bf16 conversion (grid-stride)
// ---------------------------------------------------------------------------
__global__ void cvt_kernel(const float* __restrict__ src, bf16* __restrict__ dst,
                           int n) {
  int i = blockIdx.x * blockDim.x + threadIdx.x;
  int stride = gridDim.x * blockDim.x;
  for (; i < n; i += stride) dst[i] = (bf16)src[i];
}

// ---------------------------------------------------------------------------
// Tiled bf16 WMMA GEMM:  Y[m,n] = sum_k A[m,k] * W[n,k]   (i.e. A @ W^T)
// Block tile 128x128, K-step 32, 8 waves (4x2), wave tile 32x64.
// Global -> LDS staging uses async-to-LDS (ASYNCcnt).
// mode 0: bf16 -> [B,H,S,DK] (Q,K);  mode 1: bf16 -> [B,H,DK,S] (V^T);
// mode 2: fp32 -> [MTOT,D_MODEL] + bias (final output)
// ---------------------------------------------------------------------------
__global__ __launch_bounds__(256) void gemm_kernel(
    const bf16* __restrict__ A, const bf16* __restrict__ W,
    bf16* __restrict__ outB, float* __restrict__ outF,
    const float* __restrict__ bias, int mode)
{
  __shared__ __align__(16) bf16 As[128 * 40];   // row stride 40 bf16 = 80 B
  __shared__ __align__(16) bf16 Bs[128 * 40];

  const int tid  = threadIdx.x;
  const int lane = tid & 31;
  const int wave = tid >> 5;
  const int g    = lane >> 4;   // half-wave index
  const int ln   = lane & 15;
  const int m0   = blockIdx.x * 128;
  const int n0   = blockIdx.y * 128;
  const int wm   = wave >> 1;   // 0..3
  const int wn   = wave & 1;    // 0..1

  // cooperative-load geometry: 128 rows x 4 chunks of 8 bf16 per tile
  const int row1 = tid >> 2, ck = tid & 3;   // chunks [0,256)
  const int row2 = row1 + 64;                // chunks [256,512)
  const unsigned lA1 = lds_off(&As[row1 * 40 + ck * 8]);
  const unsigned lA2 = lds_off(&As[row2 * 40 + ck * 8]);
  const unsigned lB1 = lds_off(&Bs[row1 * 40 + ck * 8]);
  const unsigned lB2 = lds_off(&Bs[row2 * 40 + ck * 8]);

  v8f acc[2][4];
#pragma unroll
  for (int mi = 0; mi < 2; ++mi)
#pragma unroll
    for (int ni = 0; ni < 4; ++ni)
#pragma unroll
      for (int j = 0; j < 8; ++j) acc[mi][ni][j] = 0.0f;

  for (int kt = 0; kt < D_MODEL / 32; ++kt) {
    async_ld_b128(lA1, &A[(size_t)(m0 + row1) * D_MODEL + kt * 32 + ck * 8]);
    async_ld_b128(lA2, &A[(size_t)(m0 + row2) * D_MODEL + kt * 32 + ck * 8]);
    async_ld_b128(lB1, &W[(size_t)(n0 + row1) * D_MODEL + kt * 32 + ck * 8]);
    async_ld_b128(lB2, &W[(size_t)(n0 + row2) * D_MODEL + kt * 32 + ck * 8]);
    wait_asynccnt0();
    __syncthreads();

    // A fragment: lane holds row m=ln, K at {8g..8g+7} and {16+8g..23+8g}
    v16bf af[2];
#pragma unroll
    for (int mi = 0; mi < 2; ++mi) {
      const bf16* r = &As[(wm * 32 + mi * 16 + ln) * 40];
      af[mi] = cat8(*(const v8bf*)(r + 8 * g), *(const v8bf*)(r + 16 + 8 * g));
    }
    // B fragment: lane holds col n=ln, K at {8g..8g+7} and {16+8g..23+8g}
    v16bf bfr[4];
#pragma unroll
    for (int ni = 0; ni < 4; ++ni) {
      const bf16* r = &Bs[(wn * 64 + ni * 16 + ln) * 40];
      bfr[ni] = cat8(*(const v8bf*)(r + 8 * g), *(const v8bf*)(r + 16 + 8 * g));
    }
#pragma unroll
    for (int mi = 0; mi < 2; ++mi)
#pragma unroll
      for (int ni = 0; ni < 4; ++ni)
        acc[mi][ni] = wmma_bf16(af[mi], bfr[ni], acc[mi][ni]);
    __syncthreads();
  }

  // epilogue: C layout -> row m = j + 8g, col n = ln (per 16x16 tile)
#pragma unroll
  for (int mi = 0; mi < 2; ++mi)
#pragma unroll
    for (int ni = 0; ni < 4; ++ni) {
      int n = n0 + wn * 64 + ni * 16 + ln;
      if (mode == 2) {
        float bv = bias[n];
#pragma unroll
        for (int j = 0; j < 8; ++j) {
          int m = m0 + wm * 32 + mi * 16 + j + 8 * g;
          outF[(size_t)m * D_MODEL + n] = acc[mi][ni][j] + bv;
        }
      } else {
        int h = n >> 6, d = n & 63;
#pragma unroll
        for (int j = 0; j < 8; ++j) {
          int m = m0 + wm * 32 + mi * 16 + j + 8 * g;
          int b = m >> 11, s = m & (SEQ - 1);
          bf16 hv = (bf16)acc[mi][ni][j];
          if (mode == 0)
            outB[(((size_t)(b * NHEADS + h)) * SEQ + s) * DK + d] = hv;
          else
            outB[(((size_t)(b * NHEADS + h)) * DK + d) * SEQ + s] = hv;
        }
      }
    }
}

// ---------------------------------------------------------------------------
// Flash attention: one (b,h) x 128-query block per workgroup; 8 waves,
// each wave owns 16 query rows. Stream 32 keys/iter with online softmax.
// K/V tiles are staged via the Tensor Data Mover (TENSORcnt), with TDM LDS
// padding reproducing the bank-friendly strides (72 / 40 bf16).
// Q,K: [B,H,S,DK] bf16; Vt: [B,H,DK,S] bf16; O: [MTOT, D_MODEL] bf16.
// ---------------------------------------------------------------------------
__global__ __launch_bounds__(256) void flash_kernel(
    const bf16* __restrict__ Q, const bf16* __restrict__ K,
    const bf16* __restrict__ Vt, bf16* __restrict__ O)
{
  __shared__ __align__(16) bf16 Ks [32 * 72];      // 32 keys x 64 d (pad->144B rows)
  __shared__ __align__(16) bf16 Vts[64 * 40];      // 64 d x 32 keys (pad->80B rows)
  __shared__ __align__(16) bf16 Ps [8 * 16 * 40];  // per-wave P: 16 q x 32 keys

  const int tid  = threadIdx.x;
  const int lane = tid & 31;
  const int wave = tid >> 5;
  const int g    = lane >> 4;
  const int ln   = lane & 15;
  const int bh   = blockIdx.x >> 4;                // b*16 + h  (0..31)
  const int qblk = blockIdx.x & 15;
  const size_t qkbase = (size_t)bh * SEQ * DK;
  const size_t vbase  = (size_t)bh * DK * SEQ;
  const int q0 = qblk * 128 + wave * 16;

  // Q fragments (two K-steps over d) straight from global, A-layout
  v16bf qf[2];
#pragma unroll
  for (int s2 = 0; s2 < 2; ++s2) {
    const bf16* r = Q + qkbase + (size_t)(q0 + ln) * DK + s2 * 32;
    qf[s2] = cat8(*(const v8bf*)(r + 8 * g), *(const v8bf*)(r + 16 + 8 * g));
  }

  float m_run[8], l_run[8];
  v8f acc[4];
#pragma unroll
  for (int j = 0; j < 8; ++j) { m_run[j] = -1e30f; l_run[j] = 0.0f; }
#pragma unroll
  for (int t = 0; t < 4; ++t)
#pragma unroll
    for (int j = 0; j < 8; ++j) acc[t][j] = 0.0f;

  const float scale = 0.125f;  // 1/sqrt(64)
  const unsigned ldsK = lds_off(&Ks[0]);
  const unsigned ldsV = lds_off(&Vts[0]);

  for (int kb = 0; kb < SEQ / 32; ++kb) {
    // ---- TDM: one wave DMAs both tiles into LDS ----
    if (wave == 0) {
      // K tile: [32 keys x 64 d] from [2048 x 64]; rows 128B + 16B pad
      u32x4 g0k = tdm_group0(ldsK, K + qkbase + (size_t)kb * 32 * DK);
      u32x8 g1k = tdm_group1(/*tensor_d0=*/DK, /*tensor_d1=*/SEQ,
                             /*tile_d0=*/DK, /*tile_d1=*/32,
                             /*stride0=*/DK, /*pad_int=*/4, /*pad_amt=*/3);
      tdm_load_2d(g0k, g1k);
      // V tile: [64 d x 32 keys] from [64 x 2048]; rows 64B + 16B pad
      u32x4 g0v = tdm_group0(ldsV, Vt + vbase + (size_t)kb * 32);
      u32x8 g1v = tdm_group1(/*tensor_d0=*/SEQ, /*tensor_d1=*/DK,
                             /*tile_d0=*/32, /*tile_d1=*/DK,
                             /*stride0=*/SEQ, /*pad_int=*/3, /*pad_amt=*/3);
      tdm_load_2d(g0v, g1v);
      wait_tensorcnt0();
    }
    __syncthreads();

    // scores: c0 = keys 0-15, c1 = keys 16-31 (16 q rows each)
    v8f c0, c1;
#pragma unroll
    for (int j = 0; j < 8; ++j) { c0[j] = 0.0f; c1[j] = 0.0f; }
#pragma unroll
    for (int s2 = 0; s2 < 2; ++s2) {
      const bf16* r0 = &Ks[ln * 72 + s2 * 32];
      v16bf b0 = cat8(*(const v8bf*)(r0 + 8 * g), *(const v8bf*)(r0 + 16 + 8 * g));
      c0 = wmma_bf16(qf[s2], b0, c0);
      const bf16* r1 = &Ks[(16 + ln) * 72 + s2 * 32];
      v16bf b1 = cat8(*(const v8bf*)(r1 + 8 * g), *(const v8bf*)(r1 + 16 + 8 * g));
      c1 = wmma_bf16(qf[s2], b1, c1);
    }

    // online softmax: row m = j + 8g lives in one VGPR across the 16-lane half
    float alpha[8];
#pragma unroll
    for (int j = 0; j < 8; ++j) {
      float s0 = c0[j] * scale, s1 = c1[j] * scale;
      c0[j] = s0; c1[j] = s1;
      float v = fmaxf(s0, s1);
      v = fmaxf(v, __shfl_xor(v, 1, 32));
      v = fmaxf(v, __shfl_xor(v, 2, 32));
      v = fmaxf(v, __shfl_xor(v, 4, 32));
      v = fmaxf(v, __shfl_xor(v, 8, 32));
      float mn = fmaxf(m_run[j], v);
      alpha[j] = __expf(m_run[j] - mn);
      m_run[j] = mn;
    }
#pragma unroll
    for (int j = 0; j < 8; ++j) {
      c0[j] = __expf(c0[j] - m_run[j]);
      c1[j] = __expf(c1[j] - m_run[j]);
      float sum = c0[j] + c1[j];
      sum += __shfl_xor(sum, 1, 32);
      sum += __shfl_xor(sum, 2, 32);
      sum += __shfl_xor(sum, 4, 32);
      sum += __shfl_xor(sum, 8, 32);
      l_run[j] = l_run[j] * alpha[j] + sum;
    }
#pragma unroll
    for (int t = 0; t < 4; ++t)
#pragma unroll
      for (int j = 0; j < 8; ++j) acc[t][j] *= alpha[j];

    // P: C-layout -> A-layout via per-wave LDS round trip (bf16)
    bf16* pbuf = &Ps[wave * 16 * 40];
#pragma unroll
    for (int j = 0; j < 8; ++j) {
      int m = j + 8 * g;
      pbuf[m * 40 + ln]      = (bf16)c0[j];
      pbuf[m * 40 + 16 + ln] = (bf16)c1[j];
    }
    const bf16* pr = &pbuf[ln * 40];
    v16bf pf = cat8(*(const v8bf*)(pr + 8 * g), *(const v8bf*)(pr + 16 + 8 * g));

    // O += P @ V  (K=32 keys in one WMMA; 4 d-tiles of 16)
#pragma unroll
    for (int t = 0; t < 4; ++t) {
      const bf16* vr = &Vts[(t * 16 + ln) * 40];
      v16bf vf = cat8(*(const v8bf*)(vr + 8 * g), *(const v8bf*)(vr + 16 + 8 * g));
      acc[t] = wmma_bf16(pf, vf, acc[t]);
    }
    __syncthreads();
  }

  // normalize and write merged-head output: O[b*S + s, h*64 + d]
  const int h = bh & 15, b = bh >> 4;
#pragma unroll
  for (int t = 0; t < 4; ++t) {
    int d = t * 16 + ln;
#pragma unroll
    for (int j = 0; j < 8; ++j) {
      int srow = q0 + j + 8 * g;
      O[((size_t)b * SEQ + srow) * D_MODEL + h * DK + d] =
          (bf16)(acc[t][j] / l_run[j]);
    }
  }
}

// ---------------------------------------------------------------------------
// launch
// ---------------------------------------------------------------------------
extern "C" void kernel_launch(void* const* d_in, const int* in_sizes, int n_in,
                              void* d_out, int out_size, void* d_ws, size_t ws_size,
                              hipStream_t stream) {
  const float* x  = (const float*)d_in[0];
  const float* Wq = (const float*)d_in[1];
  const float* Wk = (const float*)d_in[2];
  const float* Wv = (const float*)d_in[3];
  const float* Wo = (const float*)d_in[4];
  const float* bo = (const float*)d_in[5];
  float* out = (float*)d_out;

  char* w = (char*)d_ws;
  const size_t MB = (size_t)1 << 20;
  bf16* XH  = (bf16*)(w +  0 * MB);   // [4096,1024]   8 MB
  bf16* WQH = (bf16*)(w +  8 * MB);   // [1024,1024]   2 MB
  bf16* WKH = (bf16*)(w + 10 * MB);
  bf16* WVH = (bf16*)(w + 12 * MB);
  bf16* WOH = (bf16*)(w + 14 * MB);
  bf16* QH  = (bf16*)(w + 16 * MB);   // [B,H,S,64]    8 MB
  bf16* KH  = (bf16*)(w + 24 * MB);   // [B,H,S,64]    8 MB
  bf16* VT  = (bf16*)(w + 32 * MB);   // [B,H,64,S]    8 MB
  bf16* OH  = (bf16*)(w + 40 * MB);   // [4096,1024]   8 MB  -> total 48 MB

  cvt_kernel<<<1024, 256, 0, stream>>>(x,  XH,  MTOT * D_MODEL);
  cvt_kernel<<<512,  256, 0, stream>>>(Wq, WQH, D_MODEL * D_MODEL);
  cvt_kernel<<<512,  256, 0, stream>>>(Wk, WKH, D_MODEL * D_MODEL);
  cvt_kernel<<<512,  256, 0, stream>>>(Wv, WVH, D_MODEL * D_MODEL);
  cvt_kernel<<<512,  256, 0, stream>>>(Wo, WOH, D_MODEL * D_MODEL);

  dim3 gg(MTOT / 128, D_MODEL / 128);  // 32 x 8
  gemm_kernel<<<gg, 256, 0, stream>>>(XH, WQH, QH, nullptr, nullptr, 0);
  gemm_kernel<<<gg, 256, 0, stream>>>(XH, WKH, KH, nullptr, nullptr, 0);
  gemm_kernel<<<gg, 256, 0, stream>>>(XH, WVH, VT, nullptr, nullptr, 1);

  flash_kernel<<<NBATCH * NHEADS * (SEQ / 128), 256, 0, stream>>>(QH, KH, VT, OH);

  gemm_kernel<<<gg, 256, 0, stream>>>(OH, WOH, nullptr, out, bo, 2);
}
